// RNN_Layer_7404523618955
// MI455X (gfx1250) — matrix operand order repaired and checked by reading, not compiled, not run on hardware
//
#include <hip/hip_runtime.h>
#include <math.h>

// ---------------------------------------------------------------------------
// RNN layer for MI455X (gfx1250, wave32, WMMA).
//   out = ((x@Wc + bc) -> LSTM(rec_kernel) -> @w_out + b_out)
// Wc = w_in@kernel fused offline (no nonlinearity between the projections).
// All GEMMs: v_wmma_f32_16x16x32_bf16, f32 accumulation.
// LSTM: 2 persistent WGs, gates register-resident (gate-interleaved N-tile
// ownership), cell state in VGPRs, h double-buffered in LDS (1 barrier/step),
// low-K half of rec_kernel cached in LDS (CDNA5 320KB/WGP).
// ---------------------------------------------------------------------------

typedef __attribute__((ext_vector_type(16))) __bf16 v16bf;
typedef __attribute__((ext_vector_type(8)))  __bf16 v8bf;
typedef __attribute__((ext_vector_type(8)))  float  v8f;

#define DEVINL static __device__ __forceinline__

constexpr int Bz = 32, Tz = 1024, Cz = 512, Dz = 256, Uz = 256, G4 = 1024;
constexpr int ROWS = Bz * Tz;   // 32768

// Swizzled B-fragment layout for V_WMMA_F32_16X16X32_BF16.
// Fragment (kt,nt): k in [32kt,32kt+32), n in [16nt,16nt+16).
// lanes 0-15 hold K=0..15 of column n=lane; lanes 16-31 hold K=16..31.
// Stored lane-major: frag*512 + lane*16 + slot -> 32B contiguous per lane.
DEVINL size_t swz(int k, int n, int KT) {
  int kt = k >> 5, kl = k & 31, nt = n >> 4, nl = n & 15;
  int lane = ((kl >> 4) << 4) | nl;
  int slot = kl & 15;
  return ((((size_t)nt * KT + kt) * 32) + lane) * 16 + slot;
}

DEVINL v16bf cat8(v8bf lo, v8bf hi) {
  union { struct { v8bf lo, hi; } p; v16bf v; } u;
  u.p.lo = lo; u.p.hi = hi;
  return u.v;
}

DEVINL float sigm(float x) { return 1.0f / (1.0f + __expf(-x)); }

// A-fragment (16x32 bf16) from LDS row-major tile, row stride in elements.
// lane<16 -> row=lane, K 0..15 ; lane>=16 -> row=lane-16, K 16..31
DEVINL v16bf afrag_lds(const __bf16* base, int stride, int lane) {
  const __bf16* rp = base + (lane & 15) * stride + ((lane >> 4) << 4);
  return cat8(*(const v8bf*)rp, *(const v8bf*)(rp + 8));
}

// ==========================================================================
// Prep 1: Wc = w_in @ kernel (512x1024) swizzled bf16; bc = b_in@kernel+bias
// ==========================================================================
__global__ void prep_wc(const float* __restrict__ w_in, const float* __restrict__ kern,
                        const float* __restrict__ b_in, const float* __restrict__ bias,
                        __bf16* __restrict__ wc, float* __restrict__ bc) {
  int idx = blockIdx.x * blockDim.x + threadIdx.x;   // 512*1024 threads
  int n = idx & (G4 - 1);
  int m = idx >> 10;
  if (m >= Cz) return;
  float acc = 0.f;
  for (int d = 0; d < Dz; ++d) acc += w_in[m * Dz + d] * kern[d * G4 + n];
  wc[swz(m, n, Cz >> 5)] = (__bf16)acc;
  if (m == 0) {
    float bacc = bias[n];
    for (int d = 0; d < Dz; ++d) bacc += b_in[d] * kern[d * G4 + n];
    bc[n] = bacc;
  }
}

// ==========================================================================
// Prep 2: generic f32 [K][N] -> swizzled bf16 B-fragment buffer
// ==========================================================================
__global__ void swizzle_w(const float* __restrict__ src, __bf16* __restrict__ dst,
                          int K, int N) {
  int idx = blockIdx.x * blockDim.x + threadIdx.x;
  if (idx >= K * N) return;
  int k = idx / N, n = idx - k * N;
  dst[swz(k, n, K >> 5)] = (__bf16)src[idx];
}

// ==========================================================================
// GEMM 1: xz[32768,1024] = x[32768,512] @ Wc + bc    (bf16 out, f32 accum)
// Block = 512 thr (16 waves); tile 32 rows x 1024 cols; wave owns 4 N-tiles.
// ==========================================================================
__global__ __launch_bounds__(512) void xz_gemm(const float* __restrict__ x,
                                               const __bf16* __restrict__ wc,
                                               const float* __restrict__ bc,
                                               __bf16* __restrict__ xz) {
  constexpr int ASTR = 40;                     // padded row stride (bank spread)
  __shared__ __align__(32) __bf16 As[32 * ASTR];
  const int tid = threadIdx.x, lane = tid & 31, wv = tid >> 5;
  const int rowBase = blockIdx.x * 32;
  const int ntBase = wv * 4;
  const v8f vzero = {};
  v8f acc[2][4];
#pragma unroll
  for (int mt = 0; mt < 2; ++mt)
#pragma unroll
    for (int j = 0; j < 4; ++j) acc[mt][j] = vzero;

  const int sr = tid >> 4;             // staging row 0..31
  const int skk = (tid & 15) * 2;      // staging col pair (even)

  for (int kt = 0; kt < 16; ++kt) {
    __syncthreads();
    {  // vectorized stage: float2 -> packed bf16x2 (v_cvt_pk_bf16_f32)
      float2 v = *(const float2*)(x + (size_t)(rowBase + sr) * Cz + kt * 32 + skk);
      union { __bf16 h[2]; unsigned u; } p;
      p.h[0] = (__bf16)v.x; p.h[1] = (__bf16)v.y;
      *(unsigned*)&As[sr * ASTR + skk] = p.u;
    }
    __syncthreads();
    v16bf a0 = afrag_lds(&As[0], ASTR, lane);
    v16bf a1 = afrag_lds(&As[16 * ASTR], ASTR, lane);
#pragma unroll
    for (int j = 0; j < 4; ++j) {
      const __bf16* bp = wc + ((((size_t)(ntBase + j)) * 16 + kt) * 32 + lane) * 16;
      v16bf b = *(const v16bf*)bp;
      acc[0][j] = __builtin_amdgcn_wmma_f32_16x16x32_bf16(false, a0, false, b,
                                                          (short)0, acc[0][j], false, false);
      acc[1][j] = __builtin_amdgcn_wmma_f32_16x16x32_bf16(false, a1, false, b,
                                                          (short)0, acc[1][j], false, false);
    }
  }
  // C/D layout: VGPR r -> lanes0-15 (M=r,N=lane), lanes16-31 (M=r+8,N=lane-16)
#pragma unroll
  for (int mt = 0; mt < 2; ++mt)
#pragma unroll
    for (int j = 0; j < 4; ++j) {
      int col = (ntBase + j) * 16 + (lane & 15);
      float bcv = bc[col];
#pragma unroll
      for (int r = 0; r < 8; ++r) {
        int row = rowBase + mt * 16 + ((lane < 16) ? r : r + 8);
        xz[(size_t)row * G4 + col] = (__bf16)(acc[mt][j][r] + bcv);
      }
    }
}

// ==========================================================================
// LSTM scan: 2 persistent WGs (16 batch rows each).
// Wave w owns N-tiles {w, 16+w, 32+w, 48+w} -> acc[0..3] are gates i,f,g,o
// for the SAME (row,u) slab => gate math + cell state fully in registers.
// h double-buffered in LDS -> exactly one barrier per timestep.
// Low-K half of rec (kt 0..3) cached in LDS (256 KB); high half from L2.
// ==========================================================================
__global__ __launch_bounds__(512) void lstm_scan(const __bf16* __restrict__ xz,
                                                 const __bf16* __restrict__ rec,
                                                 __bf16* __restrict__ hs) {
  constexpr int HSTR = 264;                    // conflict-free b128 A-frag reads
  __shared__ __align__(32) __bf16 Hbuf[2][16 * HSTR];   // ~16.9 KB
  __shared__ __align__(32) __bf16 RecLo[256 * 512];     // 256 frags = 256 KB
  const int tid = threadIdx.x, lane = tid & 31, wv = tid >> 5;
  const int batchBase = blockIdx.x * 16;
  const v8f vzero = {};

  // cache kt 0..3 of all 64 nt fragments: global frag (nt*8+kt) -> LDS (nt*4+kt)
  for (int f = wv; f < 256; f += 16) {
    int nt = f >> 2, kt = f & 3;
    const __bf16* src = rec + (((size_t)nt * 8 + kt) * 32 + lane) * 16;
    *(v16bf*)&RecLo[((nt * 4 + kt) * 32 + lane) * 16] = *(const v16bf*)src;
  }
  for (int e = tid; e < 2 * 16 * HSTR; e += 512) ((__bf16*)Hbuf)[e] = (__bf16)0.f;

  float cr[8];                                // cell state, register-resident
#pragma unroll
  for (int r = 0; r < 8; ++r) cr[r] = 0.f;

  const int ucol = wv * 16 + (lane & 15);     // owned unit column
  __syncthreads();

  for (int t = 0; t < Tz; ++t) {
    const __bf16* Hrd = Hbuf[t & 1];
    __bf16* Hwr = Hbuf[(t & 1) ^ 1];
    // prefetch this step's xz rows (16 rows x 2KB) behind the GEMM
    if (tid < 256) {
      int r = tid >> 4, l = tid & 15;
      __builtin_prefetch(xz + ((size_t)(batchBase + r) * Tz + t) * G4 + l * 64, 0, 0);
    }
    v8f acc[4];
#pragma unroll
    for (int j = 0; j < 4; ++j) acc[j] = vzero;

#pragma unroll
    for (int kt = 0; kt < 4; ++kt) {          // B from LDS
      v16bf a = afrag_lds(Hrd + kt * 32, HSTR, lane);
#pragma unroll
      for (int j = 0; j < 4; ++j) {
        int nt = j * 16 + wv;
        v16bf b = *(const v16bf*)&RecLo[((nt * 4 + kt) * 32 + lane) * 16];
        acc[j] = __builtin_amdgcn_wmma_f32_16x16x32_bf16(false, a, false, b,
                                                         (short)0, acc[j], false, false);
      }
    }
#pragma unroll
    for (int kt = 4; kt < 8; ++kt) {          // B from L2
      v16bf a = afrag_lds(Hrd + kt * 32, HSTR, lane);
#pragma unroll
      for (int j = 0; j < 4; ++j) {
        int nt = j * 16 + wv;
        const __bf16* bp = rec + (((size_t)nt * 8 + kt) * 32 + lane) * 16;
        v16bf b = *(const v16bf*)bp;
        acc[j] = __builtin_amdgcn_wmma_f32_16x16x32_bf16(false, a, false, b,
                                                         (short)0, acc[j], false, false);
      }
    }
    // gates: acc[0..3][r] = i,f,g,o pre-activations at (row m, col ucol)
#pragma unroll
    for (int r = 0; r < 8; ++r) {
      int m = (lane < 16) ? r : r + 8;
      size_t g = (size_t)(batchBase + m) * Tz + t;
      const __bf16* xr = xz + g * G4 + ucol;
      float zi = acc[0][r] + (float)xr[0];
      float zf = acc[1][r] + (float)xr[256];
      float zg = acc[2][r] + (float)xr[512];
      float zo = acc[3][r] + (float)xr[768];
      float iv = sigm(zi), fv = sigm(zf), gv = tanhf(zg), ov = sigm(zo);
      float c = fv * cr[r] + iv * gv;
      cr[r] = c;
      float h = ov * tanhf(c);
      Hwr[m * HSTR + ucol] = (__bf16)h;
      hs[g * Uz + ucol] = (__bf16)h;
    }
    __syncthreads();   // single barrier: Hwr complete before next step reads it
  }
}

// ==========================================================================
// GEMM 2: out[32768,256] = hs[32768,256] @ w_out + b_out   (f32 out)
// ==========================================================================
__global__ __launch_bounds__(512) void out_gemm(const __bf16* __restrict__ hs,
                                                const __bf16* __restrict__ wo,
                                                const float* __restrict__ b_out,
                                                float* __restrict__ out) {
  constexpr int ASTR = 40;
  __shared__ __align__(32) __bf16 As[32 * ASTR];
  const int tid = threadIdx.x, lane = tid & 31, wv = tid >> 5;
  const int rowBase = blockIdx.x * 32;
  const int nt = wv;
  const v8f vzero = {};
  v8f acc[2] = {vzero, vzero};

  const int sr = tid >> 4;
  const int skk = (tid & 15) * 2;

  for (int kt = 0; kt < 8; ++kt) {
    __syncthreads();
    *(unsigned*)&As[sr * ASTR + skk] =
        *(const unsigned*)(hs + (size_t)(rowBase + sr) * Uz + kt * 32 + skk);
    __syncthreads();
    v16bf a0 = afrag_lds(&As[0], ASTR, lane);
    v16bf a1 = afrag_lds(&As[16 * ASTR], ASTR, lane);
    v16bf b = *(const v16bf*)(wo + (((size_t)nt * 8 + kt) * 32 + lane) * 16);
    acc[0] = __builtin_amdgcn_wmma_f32_16x16x32_bf16(false, a0, false, b,
                                                     (short)0, acc[0], false, false);
    acc[1] = __builtin_amdgcn_wmma_f32_16x16x32_bf16(false, a1, false, b,
                                                     (short)0, acc[1], false, false);
  }
  int col = nt * 16 + (lane & 15);
  float bo = b_out[col];
#pragma unroll
  for (int mt = 0; mt < 2; ++mt)
#pragma unroll
    for (int r = 0; r < 8; ++r) {
      int row = rowBase + mt * 16 + ((lane < 16) ? r : r + 8);
      out[(size_t)row * Uz + col] = acc[mt][r] + bo;
    }
}

// ==========================================================================
extern "C" void kernel_launch(void* const* d_in, const int* in_sizes, int n_in,
                              void* d_out, int out_size, void* d_ws, size_t ws_size,
                              hipStream_t stream) {
  (void)in_sizes; (void)n_in; (void)out_size; (void)ws_size;
  const float* x     = (const float*)d_in[0];
  const float* w_in  = (const float*)d_in[1];
  const float* b_in  = (const float*)d_in[2];
  const float* kern  = (const float*)d_in[3];
  const float* rec_k = (const float*)d_in[4];
  const float* bias  = (const float*)d_in[5];
  const float* w_out = (const float*)d_in[6];
  const float* b_out = (const float*)d_in[7];
  float* out = (float*)d_out;

  char* ws = (char*)d_ws;
  // workspace layout (32B aligned), ~82 MB total
  __bf16* wc  = (__bf16*)(ws);                                   // 1 MB
  float*  bc  = (float*)(ws + 1048576);                          // 4 KB
  __bf16* rec = (__bf16*)(ws + 1052672);                         // 512 KB
  __bf16* wo  = (__bf16*)(ws + 1576960);                         // 128 KB
  __bf16* xz  = (__bf16*)(ws + 1708032);                         // 64 MB
  __bf16* hs  = (__bf16*)(ws + 68816896);                        // 16 MB

  prep_wc<<<(Cz * G4) / 256, 256, 0, stream>>>(w_in, kern, b_in, bias, wc, bc);
  swizzle_w<<<(Dz * G4 + 255) / 256, 256, 0, stream>>>(rec_k, rec, Dz, G4);
  swizzle_w<<<(Uz * Uz + 255) / 256, 256, 0, stream>>>(w_out, wo, Uz, Uz);
  xz_gemm<<<ROWS / 32, 512, 0, stream>>>(x, wc, bc, xz);
  lstm_scan<<<2, 512, 0, stream>>>(xz, rec, hs);
  out_gemm<<<ROWS / 32, 512, 0, stream>>>(hs, wo, b_out, out);
}
